// MultiheadAttention_75883482186218
// MI455X (gfx1250) — matrix operand order repaired and checked by reading, not compiled, not run on hardware
//
#include <hip/hip_runtime.h>
#include <hip/hip_bf16.h>
#include <math.h>

typedef __bf16 bf16;
typedef __attribute__((ext_vector_type(16))) __bf16 v16bf;
typedef __attribute__((ext_vector_type(8)))  float  v8f;
typedef __attribute__((ext_vector_type(4)))  unsigned int u32x4;
typedef __attribute__((ext_vector_type(8)))  int i32x8;
typedef __attribute__((ext_vector_type(4)))  int i32x4;

union V16 { v16bf v; u32x4 q[2]; };

#define BATCH  2
#define SLEN   2048
#define DMODEL 1024
#define NHEAD  16
#define HDIM   64
#define MROWS  (BATCH * SLEN)   // 4096

static __device__ __forceinline__ bf16 f2bf(float f) {
  unsigned u = __builtin_bit_cast(unsigned, f);
  u += 0x7FFFu + ((u >> 16) & 1u);          // round-to-nearest-even
  unsigned short h = (unsigned short)(u >> 16);
  return __builtin_bit_cast(bf16, h);
}

static __device__ __forceinline__ v8f vzero() {
  v8f z;
#pragma unroll
  for (int i = 0; i < 8; ++i) z[i] = 0.0f;
  return z;
}

// Low 32 bits of a generic pointer to __shared__ == byte offset in LDS
static __device__ __forceinline__ unsigned lds_off(const void* p) {
  return (unsigned)(size_t)p;
}

// ---------------------------------------------------------------------------
// TDM: async DMA of a 64(N) x 32(K) bf16 tile of W (row stride 1024 elems)
// into LDS (row-major, contiguous 4 KB). Tracked with TENSORcnt.
// ---------------------------------------------------------------------------
static __device__ __forceinline__ void tdm_load_w_tile(const bf16* gptr,
                                                       unsigned ldsoff) {
  unsigned long long ga = (unsigned long long)(size_t)gptr;
  u32x4 g0;
  g0[0] = 1u;                                           // count=1 (valid)
  g0[1] = ldsoff;                                       // lds_addr
  g0[2] = (unsigned)(ga & 0xFFFFFFFFu);                 // global_addr[31:0]
  g0[3] = (unsigned)((ga >> 32) & 0x01FFFFFFu) | (2u << 30);  // [56:32], type=2
  i32x8 g1;
  g1[0] = 0x00010000;                 // data_size=1 (2 bytes/elem)
  g1[1] = (int)(1024u << 16);         // tensor_dim0 = 1024 (bits [79:48])
  g1[2] = (int)(1024u << 16);         // tensor_dim0 hi=0 | tensor_dim1 = 1024
  g1[3] = (int)(32u  << 16);          // tensor_dim1 hi=0 | tile_dim0 = 32
  g1[4] = 64;                         // tile_dim1 = 64, tile_dim2 = 0
  g1[5] = 1024;                       // tensor_dim0_stride = 1024
  g1[6] = 0;                          // stride hi | tensor_dim1_stride lo
  g1[7] = 0;
  i32x4 gz = {0, 0, 0, 0};
#if __has_include(<hip/amd_detail/amd_gfx1250_TDM.h>)
  i32x8 gz8 = {0, 0, 0, 0, 0, 0, 0, 0};
  __builtin_amdgcn_tensor_load_to_lds(g0, g1, gz, gz, gz8, 0);
#else
  __builtin_amdgcn_tensor_load_to_lds(g0, g1, gz, gz, 0);
#endif
}

// ---------------------------------------------------------------------------
// f32 -> bf16 elementwise conversion
// ---------------------------------------------------------------------------
__global__ __launch_bounds__(256) void cvt_kernel(const float* __restrict__ in,
                                                  bf16* __restrict__ out, int n) {
  int i = blockIdx.x * 256 + threadIdx.x;
  if (i < n) out[i] = f2bf(in[i]);
}

// ---------------------------------------------------------------------------
// GEMM: C[M=4096, N=1024] = A[4096,1024] @ W[1024,1024]^T + bias
// Block = 8 waves = 128(M) x 64(N) tile; wave = 16(M) x 64(N) (4 WMMA tiles).
// W K-slab (64 rows x 32 K) is TDM-DMA'd into LDS, double-buffered; B-tiles
// are then read with ds_load_b128. A-tiles stream from global (L2-resident).
// mode 0: bf16 -> [B,H,S,HD] (Q,K)   mode 1: bf16 -> [B,H,HD,S] (V^T)
// mode 2: f32  -> row-major [M,N]    (output projection)
// ---------------------------------------------------------------------------
__global__ __launch_bounds__(256) void gemm_kernel(
    const bf16* __restrict__ A, const bf16* __restrict__ W,
    const float* __restrict__ bias, void* __restrict__ Out, int mode) {
  __shared__ bf16 shW[2][64][32];           // 2 x 4 KB double buffer

  const int K    = DMODEL;
  const int lane = threadIdx.x & 31;
  const int wave = threadIdx.x >> 5;
  const int ntb  = blockIdx.x & 15;         // 16 n-blocks of 64
  const int mtb  = blockIdx.x >> 4;         // 32 m-blocks of 128
  const int half = lane >> 4;
  const int l16  = lane & 15;
  const int row  = mtb * 128 + wave * 16 + l16;

  const bf16* Wbase = W + (size_t)(ntb * 64) * K;   // slab rows n0..n0+63

  v8f acc[4];
#pragma unroll
  for (int t = 0; t < 4; ++t) acc[t] = vzero();

  const int nk = K / 32;                    // 32 k-steps
  if (wave == 0) tdm_load_w_tile(Wbase, lds_off(&shW[0][0][0]));

  for (int i = 0; i < nk; ++i) {
    if (wave == 0) {
      if (i + 1 < nk) {
        tdm_load_w_tile(Wbase + (i + 1) * 32, lds_off(&shW[(i + 1) & 1][0][0]));
        __builtin_amdgcn_s_wait_tensorcnt(1);   // oldest (this step's) done
      } else {
        __builtin_amdgcn_s_wait_tensorcnt(0);
      }
    }
    __syncthreads();                        // publish shW[i&1] to all waves

    const int k0 = i * 32;
    // A tile 16x32: lane = row l16, 16 contiguous K at 16*half
    V16 a;
    const bf16* ap = A + (size_t)row * K + k0 + 16 * half;
    a.q[0] = *(const u32x4*)(ap);
    a.q[1] = *(const u32x4*)(ap + 8);

    const bf16(*sh)[32] = shW[i & 1];
#pragma unroll
    for (int t = 0; t < 4; ++t) {
      // B tile 32x16 from LDS: lane = column, 16 contiguous K at 16*half
      const u32x4* bp = (const u32x4*)&sh[t * 16 + l16][16 * half];
      V16 b;
      b.q[0] = bp[0];
      b.q[1] = bp[1];
      acc[t] = __builtin_amdgcn_wmma_f32_16x16x32_bf16(
          false, a.v, false, b.v, (short)0, acc[t], false, false);
    }
    __syncthreads();                        // all done reading before overwrite
  }

#pragma unroll
  for (int t = 0; t < 4; ++t) {
#pragma unroll
    for (int j = 0; j < 8; ++j) {
      int m = mtb * 128 + wave * 16 + j + 8 * half;
      int n = ntb * 64 + t * 16 + l16;
      float v = acc[t][j] + bias[n];
      if (mode == 2) {
        ((float*)Out)[(size_t)m * DMODEL + n] = v;
      } else {
        int b = m >> 11, s = m & (SLEN - 1);
        int h = n >> 6, d = n & (HDIM - 1);
        size_t idx = (mode == 0)
            ? ((((size_t)b * NHEAD + h) * SLEN + s) * HDIM + d)
            : ((((size_t)b * NHEAD + h) * HDIM + d) * SLEN + s);
        ((bf16*)Out)[idx] = f2bf(v);
      }
    }
  }
}

// ---------------------------------------------------------------------------
// Fused causal flash attention: 4 independent waves/block, one wave per
// (b, h, 16-row block).  Q,K: [B,H,S,HD]  Vt: [B,H,HD,S]  O: [B*S,D] bf16
// ---------------------------------------------------------------------------
__global__ __launch_bounds__(128) void attn_kernel(const bf16* __restrict__ Q,
                                                   const bf16* __restrict__ K,
                                                   const bf16* __restrict__ Vt,
                                                   bf16* __restrict__ O) {
  __shared__ float ldsbuf[4][16 * 32];      // per-wave P staging tile

  const int wv   = threadIdx.x >> 5;
  const int wg   = blockIdx.x * 4 + wv;     // bh*128 + rowblock
  const int rblk = wg & 127;
  const int bh   = wg >> 7;                 // 0..31
  const int b    = bh >> 4;
  const int h    = bh & 15;
  const int lane = threadIdx.x & 31;
  const int half = lane >> 4;
  const int l16  = lane & 15;
  float* lds = ldsbuf[wv];

  const bf16* Qbh = Q  + (size_t)bh * SLEN * HDIM;
  const bf16* Kbh = K  + (size_t)bh * SLEN * HDIM;
  const bf16* Vbh = Vt + (size_t)bh * HDIM * SLEN;

  // Q A-tiles for K-chunks hd=[0,32) and hd=[32,64)
  const int qrow = rblk * 16 + l16;
  V16 qa0, qa1;
  {
    const bf16* p0 = Qbh + (size_t)qrow * HDIM + 0  + 8 * half;
    const bf16* p1 = Qbh + (size_t)qrow * HDIM + 32 + 8 * half;
    qa0.q[0] = *(const u32x4*)(p0);  qa0.q[1] = *(const u32x4*)(p0 + 16);
    qa1.q[0] = *(const u32x4*)(p1);  qa1.q[1] = *(const u32x4*)(p1 + 16);
  }

  v8f acc[4];
#pragma unroll
  for (int t = 0; t < 4; ++t) acc[t] = vzero();
  float rmax[8], rsum[8];
#pragma unroll
  for (int j = 0; j < 8; ++j) { rmax[j] = -__builtin_inff(); rsum[j] = 0.0f; }

  const int nkeys = rblk * 16 + 16;
  const int njt   = (nkeys + 31) >> 5;      // 32-key steps
  const float scale = 0.125f;               // 1/sqrt(64)

  for (int jt = 0; jt < njt; ++jt) {
    const int k0 = jt * 32;

    if (jt + 1 < njt) {                     // prefetch next K/V tiles
      __builtin_prefetch(Kbh + (size_t)(k0 + 32 + lane) * HDIM, 0, 0);
      __builtin_prefetch(Vbh + (size_t)lane * SLEN + k0 + 32, 0, 0);
      __builtin_prefetch(Vbh + (size_t)(32 + lane) * SLEN + k0 + 32, 0, 0);
    }

    // ---- scores: two 16x16 tiles covering keys k0..k0+15 and k0+16..k0+31
    v8f s[2];
#pragma unroll
    for (int t = 0; t < 2; ++t) {
      int keyn = k0 + t * 16 + l16;
      const bf16* kp = Kbh + (size_t)keyn * HDIM + 8 * half;
      V16 kb0, kb1;
      kb0.q[0] = *(const u32x4*)(kp);       kb0.q[1] = *(const u32x4*)(kp + 16);
      kb1.q[0] = *(const u32x4*)(kp + 32);  kb1.q[1] = *(const u32x4*)(kp + 48);
      v8f st = vzero();
      st = __builtin_amdgcn_wmma_f32_16x16x32_bf16(false, qa0.v, false, kb0.v,
                                                   (short)0, st, false, false);
      st = __builtin_amdgcn_wmma_f32_16x16x32_bf16(false, qa1.v, false, kb1.v,
                                                   (short)0, st, false, false);
      s[t] = st;
    }

    // ---- scale + causal mask (element (m,n): m = j+8*half, n = l16)
    float sv[2][8];
#pragma unroll
    for (int t = 0; t < 2; ++t) {
#pragma unroll
      for (int j = 0; j < 8; ++j) {
        int m  = j + 8 * half;
        int ig = rblk * 16 + m;
        int jg = k0 + t * 16 + l16;
        float v = s[t][j] * scale;
        sv[t][j] = (jg > ig) ? -__builtin_inff() : v;
      }
    }

    // ---- online softmax: per-row max over the 16-lane half-wave group
    float fac[8];
#pragma unroll
    for (int j = 0; j < 8; ++j) {
      float mx = fmaxf(sv[0][j], sv[1][j]);
#pragma unroll
      for (int off = 8; off >= 1; off >>= 1)
        mx = fmaxf(mx, __shfl_xor(mx, off, 16));
      float mnew = fmaxf(rmax[j], mx);
      fac[j]  = __expf(rmax[j] - mnew);     // 0 on first step (rmax = -inf)
      rmax[j] = mnew;
    }

    // ---- P = exp(s - mnew), row sums, stage P into LDS
    float psum[8];
#pragma unroll
    for (int j = 0; j < 8; ++j) psum[j] = 0.0f;
#pragma unroll
    for (int t = 0; t < 2; ++t) {
#pragma unroll
      for (int j = 0; j < 8; ++j) {
        float p = __expf(sv[t][j] - rmax[j]);
        psum[j] += p;
        lds[(j + 8 * half) * 32 + t * 16 + l16] = p;
      }
    }
#pragma unroll
    for (int j = 0; j < 8; ++j) {
      float rs = psum[j];
#pragma unroll
      for (int off = 8; off >= 1; off >>= 1)
        rs += __shfl_xor(rs, off, 16);
      rsum[j] = rsum[j] * fac[j] + rs;
    }

    // ---- rescale running accumulators
#pragma unroll
    for (int t = 0; t < 4; ++t)
#pragma unroll
      for (int j = 0; j < 8; ++j) acc[t][j] *= fac[j];

    // ---- re-layout P (LDS row-major f32) into A-matrix bf16 striping
    V16 pa;
#pragma unroll
    for (int e = 0; e < 8; ++e) {
      pa.v[e]     = f2bf(lds[l16 * 32 + 8 * half + e]);
      pa.v[e + 8] = f2bf(lds[l16 * 32 + 16 + 8 * half + e]);
    }

    // ---- acc += P(16x32) x V(32x16) for 4 head-dim tiles
#pragma unroll
    for (int t = 0; t < 4; ++t) {
      int d = t * 16 + l16;
      const bf16* vp = Vbh + (size_t)d * SLEN + k0 + 8 * half;
      V16 vb;
      vb.q[0] = *(const u32x4*)(vp);
      vb.q[1] = *(const u32x4*)(vp + 16);
      acc[t] = __builtin_amdgcn_wmma_f32_16x16x32_bf16(
          false, pa.v, false, vb.v, (short)0, acc[t], false, false);
    }
  }

  // ---- epilogue: normalize and write [B*S, D] bf16
#pragma unroll
  for (int t = 0; t < 4; ++t) {
#pragma unroll
    for (int j = 0; j < 8; ++j) {
      int m = j + 8 * half;
      float o = acc[t][j] / rsum[j];
      size_t idx = ((size_t)b * SLEN + rblk * 16 + m) * DMODEL +
                   h * HDIM + t * 16 + l16;
      O[idx] = f2bf(o);
    }
  }
}

// ---------------------------------------------------------------------------
extern "C" void kernel_launch(void* const* d_in, const int* in_sizes, int n_in,
                              void* d_out, int out_size, void* d_ws, size_t ws_size,
                              hipStream_t stream) {
  const float* x  = (const float*)d_in[0];
  const float* Wq = (const float*)d_in[1];
  const float* bq = (const float*)d_in[2];
  const float* Wk = (const float*)d_in[3];
  const float* bk = (const float*)d_in[4];
  const float* Wv = (const float*)d_in[5];
  const float* bv = (const float*)d_in[6];
  const float* Wo = (const float*)d_in[7];
  const float* bo = (const float*)d_in[8];

  const size_t MB = 1024 * 1024;
  char* w = (char*)d_ws;
  bf16* xh  = (bf16*)(w + 0 * MB);    // 8 MiB  x bf16 [4096,1024]
  bf16* wqh = (bf16*)(w + 8 * MB);    // 2 MiB each
  bf16* wkh = (bf16*)(w + 10 * MB);
  bf16* wvh = (bf16*)(w + 12 * MB);
  bf16* woh = (bf16*)(w + 14 * MB);
  bf16* qh  = (bf16*)(w + 16 * MB);   // 8 MiB  [B,H,S,HD]
  bf16* kh  = (bf16*)(w + 24 * MB);   // 8 MiB  [B,H,S,HD]
  bf16* vth = (bf16*)(w + 32 * MB);   // 8 MiB  [B,H,HD,S]
  bf16* ah  = (bf16*)(w + 40 * MB);   // 8 MiB  attn output [4096,1024]

  const int NX = MROWS * DMODEL;      // 4194304
  const int NW = DMODEL * DMODEL;     // 1048576
  cvt_kernel<<<(NX + 255) / 256, 256, 0, stream>>>(x,  xh,  NX);
  cvt_kernel<<<(NW + 255) / 256, 256, 0, stream>>>(Wq, wqh, NW);
  cvt_kernel<<<(NW + 255) / 256, 256, 0, stream>>>(Wk, wkh, NW);
  cvt_kernel<<<(NW + 255) / 256, 256, 0, stream>>>(Wv, wvh, NW);
  cvt_kernel<<<(NW + 255) / 256, 256, 0, stream>>>(Wo, woh, NW);

  // QKV projections: 32 m-blocks(128) x 16 n-blocks(64) = 512 blocks
  gemm_kernel<<<512, 256, 0, stream>>>(xh, wqh, bq, (void*)qh,  0);
  gemm_kernel<<<512, 256, 0, stream>>>(xh, wkh, bk, (void*)kh,  0);
  gemm_kernel<<<512, 256, 0, stream>>>(xh, wvh, bv, (void*)vth, 1);

  // Fused causal attention: 4096 waves, 4 per block
  attn_kernel<<<1024, 128, 0, stream>>>(qh, kh, vth, ah);

  // Output projection -> f32 d_out
  gemm_kernel<<<512, 256, 0, stream>>>(ah, woh, bo, d_out, 2);
}
